// CustomSimplexMappingAttention_72962904424590
// MI455X (gfx1250) — compile-verified, matrix operand order checked by reading, not run on hardware
//
#include <hip/hip_runtime.h>

// ---- problem constants (match reference) ----
#define BATCH 2
#define LSEQ  2048
#define DM    512
#define NH    8
#define HD    64
#define NEGV  (-1.0e9f)

typedef __bf16 bf16_t;
typedef __attribute__((ext_vector_type(16))) __bf16 v16bf;
typedef __attribute__((ext_vector_type(8)))  __bf16 v8bf;
typedef __attribute__((ext_vector_type(8)))  float  v8f;
typedef __attribute__((ext_vector_type(4)))  int    v4i;

union AFrag { v16bf v; bf16_t e[16]; };
union CFrag { v8f   v; float  e[8];  };

// ---------------------------------------------------------------------------
// CDNA5 async memory->LDS staging (ASYNCcnt-tracked), guarded so either
// toolchain compiles: fall back to VGPR round-trip if builtins are absent.
// Probe round 2 showed the builtin expects v4i* (generic) pointers.
// ---------------------------------------------------------------------------
#if __has_builtin(__builtin_amdgcn_global_load_async_to_lds_b128)
#define HAVE_ASYNC_LDS 1
__device__ inline void async_copy_b128(void* lds, void* gsrc) {
    __builtin_amdgcn_global_load_async_to_lds_b128((v4i*)gsrc, (v4i*)lds, 0, 0);
}
#else
#define HAVE_ASYNC_LDS 0
#endif

__device__ inline void wait_async_lds() {
#if __has_builtin(__builtin_amdgcn_s_wait_asynccnt)
    __builtin_amdgcn_s_wait_asynccnt(0);
#else
    asm volatile("s_wait_asynccnt 0x0" ::: "memory");
#endif
}

// ---------------------------------------------------------------------------
// fp32 -> bf16 pack
// ---------------------------------------------------------------------------
__global__ __launch_bounds__(256) void pack_bf16(const float* __restrict__ src,
                                                 bf16_t* __restrict__ dst, int n) {
    for (int i = blockIdx.x * 256 + threadIdx.x; i < n; i += gridDim.x * 256)
        dst[i] = (bf16_t)src[i];
}

// ---------------------------------------------------------------------------
// A-fragment (16x32 bf16) loader following ISA 7.12.2 16-bit striping:
// lanes 0-15 row m=lane, K = {kb..kb+7} U {kb+16..kb+23}, kb = (lane>>4)*8
// Same pattern serves B (32x16) when fed with W rows (B[k][n] = W[n][k]).
// ---------------------------------------------------------------------------
__device__ inline v16bf load_frag_rowpair(const bf16_t* base, int ld) {
    int lane = threadIdx.x & 31;
    int row  = lane & 15;
    int kb   = (lane >> 4) * 8;
    const bf16_t* p = base + row * ld + kb;
    AFrag f;
    *(v8bf*)(&f.e[0]) = *(const v8bf*)(p);
    *(v8bf*)(&f.e[8]) = *(const v8bf*)(p + 16);
    return f.v;
}

// ---------------------------------------------------------------------------
// C[M,N] = A[M,K](bf16,row-major) x W[N,K]^T (bf16, torch Linear weight layout)
// MODE 0: write bf16 in [b, h, l, hd] layout        (Q, K)
// MODE 1: write f32 row-major                       (final W_o projection)
// MODE 2: write bf16 in [b, h, hd, l] layout        (V pre-transposed per head)
// Block: 256 thr = 8 waves; block tile 64(M) x 64(N); K stepped by 32 via LDS.
// ---------------------------------------------------------------------------
template <int MODE>
__global__ __launch_bounds__(256) void gemm_wt(const bf16_t* __restrict__ A,
                                               const bf16_t* __restrict__ W,
                                               void* __restrict__ out,
                                               int M, int N, int K) {
    __shared__ bf16_t As[64 * 32];
    __shared__ bf16_t Bs[64 * 32];

    const int nblk = N / 64;
    const int m0 = (blockIdx.x / nblk) * 64;
    const int n0 = (blockIdx.x % nblk) * 64;
    const int tid  = threadIdx.x;
    const int wid  = tid >> 5;
    const int lane = tid & 31;
    const int wm = wid & 3;        // 4 M sub-tiles of 16 rows
    const int wn = wid >> 2;       // 2 N sub-tiles of 32 cols

    CFrag acc0, acc1;
    for (int i = 0; i < 8; ++i) { acc0.e[i] = 0.f; acc1.e[i] = 0.f; }

    const int r  = tid >> 2;        // 0..63
    const int cb = (tid & 3) * 8;   // 0,8,16,24

    for (int k0 = 0; k0 < K; k0 += 32) {
        bf16_t* ga = (bf16_t*)&A[(size_t)(m0 + r) * K + k0 + cb];
        bf16_t* gw = (bf16_t*)&W[(size_t)(n0 + r) * K + k0 + cb];
#if HAVE_ASYNC_LDS
        // GLOBAL_LOAD_ASYNC_TO_LDS_B128: memory->LDS, no VGPR round-trip
        async_copy_b128(&As[r * 32 + cb], ga);
        async_copy_b128(&Bs[r * 32 + cb], gw);
#else
        *(v8bf*)&As[r * 32 + cb] = *(const v8bf*)ga;
        *(v8bf*)&Bs[r * 32 + cb] = *(const v8bf*)gw;
#endif
        if (k0 + 32 < K) {  // far prefetch of next K-slice: global_prefetch_b8
            __builtin_prefetch(ga + 32, 0, 1);
            __builtin_prefetch(gw + 32, 0, 1);
        }
#if HAVE_ASYNC_LDS
        wait_async_lds();
#endif
        __syncthreads();

        v16bf a  = load_frag_rowpair(&As[wm * 16 * 32], 32);
        v16bf b0 = load_frag_rowpair(&Bs[(wn * 32 + 0) * 32], 32);
        v16bf b1 = load_frag_rowpair(&Bs[(wn * 32 + 16) * 32], 32);
        acc0.v = __builtin_amdgcn_wmma_f32_16x16x32_bf16(false, a, false, b0,
                                                         (short)0, acc0.v, false, false);
        acc1.v = __builtin_amdgcn_wmma_f32_16x16x32_bf16(false, a, false, b1,
                                                         (short)0, acc1.v, false, false);
        __syncthreads();
    }

    // C layout: VGPR r -> M = r (+8 for lanes 16-31), N = lane&15
    const int rhi = (lane >> 4) * 8;
    const int col = lane & 15;
    for (int rr = 0; rr < 8; ++rr) {
        int m   = m0 + wm * 16 + rr + rhi;
        int n_a = n0 + wn * 32 + col;
        int n_b = n_a + 16;
        int b = m / LSEQ, l = m % LSEQ;
        if (MODE == 0) {
            bf16_t* o = (bf16_t*)out;
            int h0 = n_a / HD, hd0 = n_a % HD;
            int h1 = n_b / HD, hd1 = n_b % HD;
            o[(((size_t)b * NH + h0) * LSEQ + l) * HD + hd0] = (bf16_t)acc0.e[rr];
            o[(((size_t)b * NH + h1) * LSEQ + l) * HD + hd1] = (bf16_t)acc1.e[rr];
        } else if (MODE == 2) {
            bf16_t* o = (bf16_t*)out;   // Vt[b, h, hd, l]
            int h0 = n_a / HD, hd0 = n_a % HD;
            int h1 = n_b / HD, hd1 = n_b % HD;
            o[(((size_t)b * NH + h0) * HD + hd0) * LSEQ + l] = (bf16_t)acc0.e[rr];
            o[(((size_t)b * NH + h1) * HD + hd1) * LSEQ + l] = (bf16_t)acc1.e[rr];
        } else {
            float* o = (float*)out;
            o[(size_t)m * N + n_a] = acc0.e[rr];
            o[(size_t)m * N + n_b] = acc1.e[rr];
        }
    }
}

// ---------------------------------------------------------------------------
// Fused causal sparsemax attention for one (b,h) x 16-query tile.
// Phase 1: scores tile [16 x width] via WMMA QK^T -> LDS f32 (causal mask)
// Phase 2: per-row sparsemax: max + 26-step bisection for tau -> bf16 probs
// Phase 3: out[16x64] = probs @ Vt via WMMA, cross-wave ds_add_f32 reduce
// LDS: 128KB f32 scores + 64KB bf16 probs + 4KB f32 out = 200704 B (<320KB WGP)
// ---------------------------------------------------------------------------
__global__ __launch_bounds__(256) void sparsemax_attn(const bf16_t* __restrict__ Q,
                                                      const bf16_t* __restrict__ Km,
                                                      const bf16_t* __restrict__ Vt,
                                                      bf16_t* __restrict__ O) {
    extern __shared__ char smem[];
    float*  scores = (float*)smem;                                  // 16 x 2048 f32
    bf16_t* attnb  = (bf16_t*)(smem + 16 * LSEQ * 4);               // 16 x 2048 bf16
    float*  outl   = (float*)(smem + 16 * LSEQ * 4 + 16 * LSEQ * 2); // 16 x 64 f32

    const int bid = blockIdx.x;
    const int qt  = bid % (LSEQ / 16);
    const int bh  = bid / (LSEQ / 16);     // b*NH + h
    const int q0  = qt * 16;
    const int tid = threadIdx.x, wid = tid >> 5, lane = tid & 31;
    const int laneRow = lane & 15;
    const int kb      = (lane >> 4) * 8;
    const int rhi     = (lane >> 4) * 8;
    const int col     = lane & 15;

    const bf16_t* Qb  = Q  + ((size_t)bh * LSEQ + q0) * HD;
    const bf16_t* Kb  = Km + (size_t)bh * LSEQ * HD;
    const bf16_t* Vtb = Vt + (size_t)bh * HD * LSEQ;   // [hd][l]

    for (int i = tid; i < 16 * HD; i += 256) outl[i] = 0.f;

    // Q fragments (A-matrix, 16x64 -> two 16x32)
    AFrag a0, a1;
    {
        const bf16_t* p = Qb + laneRow * HD + kb;
        *(v8bf*)&a0.e[0] = *(const v8bf*)(p);
        *(v8bf*)&a0.e[8] = *(const v8bf*)(p + 16);
        *(v8bf*)&a1.e[0] = *(const v8bf*)(p + 32);
        *(v8bf*)&a1.e[8] = *(const v8bf*)(p + 48);
    }

    // ---- Phase 1: scores (skip fully-masked key tiles; kt <= qt) ----
    for (int kt = wid; kt <= qt; kt += 8) {
        AFrag b0, b1;  // B[k][n] = K[key=n][hd=k]: K rows are contiguous in k
        const bf16_t* kp = Kb + (size_t)(kt * 16 + laneRow) * HD + kb;
        *(v8bf*)&b0.e[0] = *(const v8bf*)(kp);
        *(v8bf*)&b0.e[8] = *(const v8bf*)(kp + 16);
        *(v8bf*)&b1.e[0] = *(const v8bf*)(kp + 32);
        *(v8bf*)&b1.e[8] = *(const v8bf*)(kp + 48);

        CFrag c; for (int i = 0; i < 8; ++i) c.e[i] = 0.f;
        c.v = __builtin_amdgcn_wmma_f32_16x16x32_bf16(false, a0.v, false, b0.v,
                                                      (short)0, c.v, false, false);
        c.v = __builtin_amdgcn_wmma_f32_16x16x32_bf16(false, a1.v, false, b1.v,
                                                      (short)0, c.v, false, false);
        for (int r2 = 0; r2 < 8; ++r2) {
            int row  = r2 + rhi;
            int kcol = kt * 16 + col;
            float sv = c.e[r2] * 0.125f;   // 1/sqrt(64)
            scores[row * LSEQ + kcol] = (kcol <= q0 + row) ? sv : NEGV;
        }
    }
    __syncthreads();

    // ---- Phase 2: sparsemax per row (2 rows per wave) ----
    const int width = (qt + 1) * 16;
    for (int ri = 0; ri < 2; ++ri) {
        int row = wid * 2 + ri;
        const float* sr = &scores[row * LSEQ];
        int nvalid = q0 + row + 1;

        float mx = -3.0e38f;
        for (int c = lane; c < nvalid; c += 32) mx = fmaxf(mx, sr[c]);
        for (int off = 16; off > 0; off >>= 1) mx = fmaxf(mx, __shfl_xor(mx, off, 32));

        // tau in [mx-1, mx]; s(tau) = sum relu(z - tau) monotone decreasing
        float lo = mx - 1.0f, hi = mx;
        for (int it = 0; it < 26; ++it) {
            float tau = 0.5f * (lo + hi);
            float s = 0.f;
            for (int c = lane; c < nvalid; c += 32) s += fmaxf(sr[c] - tau, 0.f);
            for (int off = 16; off > 0; off >>= 1) s += __shfl_xor(s, off, 32);
            if (s >= 1.0f) lo = tau; else hi = tau;
        }
        float tau = 0.5f * (lo + hi);

        bf16_t* ar = &attnb[row * LSEQ];
        for (int c = lane; c < width; c += 32) {
            float p = (c < nvalid) ? fmaxf(sr[c] - tau, 0.f) : 0.f;
            ar[c] = (bf16_t)p;
        }
    }
    __syncthreads();

    // ---- Phase 3: out = probs @ V, partial per wave, ds_add_f32 reduce ----
    // B[k][n] = V[key=k][hd=n] = Vt[hd=n][key=k]: contiguous in k now.
    CFrag o4[4];
    for (int t = 0; t < 4; ++t)
        for (int i = 0; i < 8; ++i) o4[t].e[i] = 0.f;

    for (int k32 = wid * 32; k32 < width; k32 += 8 * 32) {
        AFrag a;
        const bf16_t* ap = &attnb[laneRow * LSEQ + k32 + kb];
        *(v8bf*)&a.e[0] = *(const v8bf*)(ap);
        *(v8bf*)&a.e[8] = *(const v8bf*)(ap + 16);
        for (int t = 0; t < 4; ++t) {
            AFrag bfr;
            const bf16_t* vp = Vtb + (size_t)(t * 16 + col) * LSEQ + k32 + kb;
            *(v8bf*)&bfr.e[0] = *(const v8bf*)(vp);
            *(v8bf*)&bfr.e[8] = *(const v8bf*)(vp + 16);
            o4[t].v = __builtin_amdgcn_wmma_f32_16x16x32_bf16(false, a.v, false, bfr.v,
                                                              (short)0, o4[t].v, false, false);
        }
    }
    for (int t = 0; t < 4; ++t)
        for (int r2 = 0; r2 < 8; ++r2)
            atomicAdd(&outl[(r2 + rhi) * HD + t * 16 + col], o4[t].e[r2]);
    __syncthreads();

    __builtin_amdgcn_s_wait_tensorcnt(0);

    // store to [b, l, h*HD+hd] (row-major [4096,512]) for the W_o GEMM
    const int b = bh / NH, h = bh % NH;
    for (int i = tid; i < 16 * HD; i += 256) {
        int row = i / HD, hd = i % HD;
        O[((size_t)b * LSEQ + q0 + row) * DM + h * HD + hd] = (bf16_t)outl[i];
    }
}

// ---------------------------------------------------------------------------
// host launcher
// ---------------------------------------------------------------------------
extern "C" void kernel_launch(void* const* d_in, const int* in_sizes, int n_in,
                              void* d_out, int out_size, void* d_ws, size_t ws_size,
                              hipStream_t stream) {
    const float* x  = (const float*)d_in[0];
    const float* Wq = (const float*)d_in[1];
    const float* Wk = (const float*)d_in[2];
    const float* Wv = (const float*)d_in[3];
    const float* Wo = (const float*)d_in[4];

    const int M  = BATCH * LSEQ;        // 4096
    const int NX = M * DM;              // x elements
    const int NW = DM * DM;             // weight elements

    char* ws = (char*)d_ws;
    bf16_t* xb  = (bf16_t*)(ws);                                // 4 MB
    bf16_t* wqb = (bf16_t*)(ws + (size_t)NX * 2);
    bf16_t* wkb = wqb + NW;
    bf16_t* wvb = wkb + NW;
    bf16_t* wob = wvb + NW;
    bf16_t* Qb  = wob + NW;                                     // [B,H,L,HD]
    bf16_t* Kb  = Qb + (size_t)M * DM;
    bf16_t* Vtb = Kb + (size_t)M * DM;                          // [B,H,HD,L]
    bf16_t* AO  = Vtb + (size_t)M * DM;                         // [M, DM]

    pack_bf16<<<2048, 256, 0, stream>>>(x,  xb,  NX);
    pack_bf16<<<256,  256, 0, stream>>>(Wq, wqb, NW);
    pack_bf16<<<256,  256, 0, stream>>>(Wk, wkb, NW);
    pack_bf16<<<256,  256, 0, stream>>>(Wv, wvb, NW);
    pack_bf16<<<256,  256, 0, stream>>>(Wo, wob, NW);

    const int gemm_blocks = (M / 64) * (DM / 64);   // 512
    gemm_wt<0><<<gemm_blocks, 256, 0, stream>>>(xb, wqb, (void*)Qb,  M, DM, DM);
    gemm_wt<0><<<gemm_blocks, 256, 0, stream>>>(xb, wkb, (void*)Kb,  M, DM, DM);
    gemm_wt<2><<<gemm_blocks, 256, 0, stream>>>(xb, wvb, (void*)Vtb, M, DM, DM);

    const size_t attn_lds = (size_t)16 * LSEQ * 4 + (size_t)16 * LSEQ * 2 + 16 * HD * 4;
    sparsemax_attn<<<BATCH * NH * (LSEQ / 16), 256, attn_lds, stream>>>(Qb, Kb, Vtb, AO);

    gemm_wt<1><<<gemm_blocks, 256, 0, stream>>>(AO, wob, d_out, M, DM, DM);
}
